// TileTask_46540265619551
// MI455X (gfx1250) — 8-bit matrix operand order repaired and checked by reading, compiled, not run on hardware
//
#include <hip/hip_runtime.h>
#include <stdint.h>

typedef __attribute__((ext_vector_type(8))) int v8i;

#define B_   16
#define CIN  128
#define COUT 256
#define H_   56
#define W_   56
#define HP   58
#define WP   58
#define OH_  28
#define OW_  28

__global__ void pack_input(const float* __restrict__ x, uint32_t* __restrict__ xp) {
    int idx = blockIdx.x * blockDim.x + threadIdx.x;
    const int total = B_ * HP * WP * (CIN / 4);
    if (idx >= total) return;
    int c4 = idx % (CIN / 4);
    int t  = idx / (CIN / 4);
    int w  = t % WP; t /= WP;
    int h  = t % HP;
    int b  = t / HP;
    uint32_t out = 0u;
    const int hh = h - 1, ww = w - 1;
    if (hh >= 0 && hh < H_ && ww >= 0 && ww < W_) {
        #pragma unroll
        for (int k = 0; k < 4; ++k) {
            int c = c4 * 4 + k;
            float v = x[((size_t)(b * CIN + c) * H_ + hh) * W_ + ww];
            v = fminf(127.f, fmaxf(-128.f, v));
            int iv = (int)v;
            out |= ((uint32_t)(uint8_t)(int8_t)iv) << (8 * k);
        }
    }
    xp[idx] = out;
}

__global__ void pack_weight(const float* __restrict__ wt, uint32_t* __restrict__ wp) {
    int idx = blockIdx.x * blockDim.x + threadIdx.x;
    const int total = 9 * 2 * COUT * 16;
    if (idx >= total) return;
    int d  = idx % 16;
    int t  = idx / 16;
    int cout  = t % COUT; t /= COUT;
    int chunk = t % 2;
    int pos   = t / 2;
    const int kh = pos / 3, kw = pos % 3;
    uint32_t out = 0u;
    #pragma unroll
    for (int k = 0; k < 4; ++k) {
        int cin = chunk * 64 + d * 4 + k;
        float v = wt[((size_t)(cout * CIN + cin) * 3 + kh) * 3 + kw];
        int iv = (int)v;
        out |= ((uint32_t)(uint8_t)(int8_t)iv) << (8 * k);
    }
    wp[idx] = out;
}

__global__ __launch_bounds__(256)
void cimu_conv_fused(const int8_t* __restrict__ xp,
                     const int8_t* __restrict__ wp,
                     const float* __restrict__ bias,
                     const float* __restrict__ merge,
                     const float* __restrict__ gather,
                     const float* __restrict__ ci, const float* __restrict__ cs,
                     const float* __restrict__ si, const float* __restrict__ ss,
                     const float* __restrict__ ai, const float* __restrict__ aS,
                     float* __restrict__ out) {
    __shared__ uint32_t lds[1280];

    const int wb   = blockIdx.x;
    const int hb   = blockIdx.y;
    const int bz   = blockIdx.z;
    const int b    = bz >> 1;
    const int cgrp = bz & 1;

    const int tid  = threadIdx.x;
    const int wave = tid >> 5;
    const int lane = tid & 31;

    const uint32_t* xsrc = (const uint32_t*)xp;
    for (int d = tid; d < 1280; d += 256) {
        int lr  = d / 320;
        int rem = d % 320;
        int lc  = rem / 32;
        int c4  = rem % 32;
        int gh  = 2 * hb + lr;
        int gw  = 8 * wb + lc;
        lds[d] = xsrc[((size_t)(b * HP + gh) * WP + gw) * 32 + c4];
    }
    __syncthreads();

    const int cout0 = (cgrp * 8 + wave) * 16;
    const int n     = lane & 15;
    const int half  = lane >> 4;
    const int am    = lane & 15;
    const int ar    = am >> 3, ac = am & 7;

    v8i acc[8];
    #pragma unroll
    for (int i = 0; i < 8; ++i) acc[i] = v8i{0,0,0,0,0,0,0,0};

    const char* ldsb = (const char*)lds;

    for (int pos = 0; pos < 9; ++pos) {
        const int kh = pos / 3, kw = pos - kh * 3;
        const int apix = ((ar + kh) * 10 + (ac + kw)) * 128;
        #pragma unroll
        for (int chunk = 0; chunk < 2; ++chunk) {
            const int abase = apix + chunk * 64 + (half ? 32 : 0);
            uint2 a0 = *(const uint2*)(ldsb + abase +  0);
            uint2 a1 = *(const uint2*)(ldsb + abase +  8);
            uint2 a2 = *(const uint2*)(ldsb + abase + 16);
            uint2 a3 = *(const uint2*)(ldsb + abase + 24);
            uint32_t araw[8] = {a0.x, a0.y, a1.x, a1.y, a2.x, a2.y, a3.x, a3.y};

            const int8_t* wbase =
                wp + ((size_t)((pos * 2 + chunk) * COUT) + (cout0 + n)) * 64 + (half ? 32 : 0);
            __builtin_prefetch(wbase + COUT * 64, 0, 3);
            uint4 b0 = *(const uint4*)(wbase);
            uint4 b1 = *(const uint4*)(wbase + 16);
            v8i bvec;
            bvec[0] = (int)b0.x; bvec[1] = (int)b0.y; bvec[2] = (int)b0.z; bvec[3] = (int)b0.w;
            bvec[4] = (int)b1.x; bvec[5] = (int)b1.y; bvec[6] = (int)b1.z; bvec[7] = (int)b1.w;

            v8i bits[8];
            #pragma unroll
            for (int i = 0; i < 8; ++i) {
                #pragma unroll
                for (int j = 0; j < 8; ++j)
                    bits[i][j] = (int)((araw[j] >> (uint32_t)i) & 0x01010101u);
            }
            __builtin_amdgcn_sched_barrier(0);

            #pragma unroll
            for (int i = 0; i < 8; ++i) {
                acc[i] = __builtin_amdgcn_wmma_i32_16x16x64_iu8(
                     false, bits[i],  true, bvec,
                    acc[i],  false,  false);
            }
        }
    }

    const int   cout   = cout0 + n;
    const float bval   = bias[cout];
    const float mult_c = ci[cout] * exp2f(cs[cout]);
    const float mult_s = si[0] * exp2f(ss[0]);
    const float mult_a = ai[0] * exp2f(aS[0]);
    const int   oh     = 2 * hb + half;

    const size_t rowoff = ((size_t)(b * COUT + cout) * H_ + oh) * W_ + 8 * wb;
    float4 mg0 = *(const float4*)(merge  + rowoff);
    float4 mg1 = *(const float4*)(merge  + rowoff + 4);
    float4 gt0 = *(const float4*)(gather + rowoff);
    float4 gt1 = *(const float4*)(gather + rowoff + 4);
    float mgv[8] = {mg0.x, mg0.y, mg0.z, mg0.w, mg1.x, mg1.y, mg1.z, mg1.w};
    float gtv[8] = {gt0.x, gt0.y, gt0.z, gt0.w, gt1.x, gt1.y, gt1.z, gt1.w};

    float vals[8];
    #pragma unroll
    for (int j = 0; j < 8; ++j) {
        int s = 0;
        #pragma unroll
        for (int i = 0; i < 8; ++i) {
            int v = acc[i][j];
            v = v > 127 ? 127 : (v < -128 ? -128 : v);
            s += (i == 7) ? (v * -128) : (v << i);
        }
        float y = (float)s + bval;
        y += mgv[j];
        y  = fminf(127.f, fmaxf(-128.f, rintf(y * mult_c)));
        y += gtv[j];
        y  = fminf(127.f, fmaxf(-128.f, rintf(y * mult_s)));
        y  = fmaxf(y, 0.f);
        y  = fminf(127.f, fmaxf(-128.f, rintf(y * mult_a)));
        vals[j] = y;
    }

    #pragma unroll
    for (int k = 0; k < 4; ++k) {
        float m2    = fmaxf(vals[2 * k], vals[2 * k + 1]);
        float other = __shfl_xor(m2, 16, 32);
        float mm    = fmaxf(m2, other);
        if (half == 0) {
            const int owp = 4 * wb + k;
            out[((size_t)(b * COUT + cout) * OH_ + hb) * OW_ + owp] = mm;
        }
    }
}

extern "C" void kernel_launch(void* const* d_in, const int* in_sizes, int n_in,
                              void* d_out, int out_size, void* d_ws, size_t ws_size,
                              hipStream_t stream) {
    const float* cast_in   = (const float*)d_in[0];
    const float* merge_in  = (const float*)d_in[1];
    const float* gather_in = (const float*)d_in[2];
    const float* conv_w    = (const float*)d_in[3];
    const float* conv_b    = (const float*)d_in[4];
    const float* ctrans_i  = (const float*)d_in[5];
    const float* ctrans_s  = (const float*)d_in[6];
    const float* strans_i  = (const float*)d_in[7];
    const float* strans_s  = (const float*)d_in[8];
    const float* atrans_i  = (const float*)d_in[9];
    const float* atrans_s  = (const float*)d_in[10];
    float* out = (float*)d_out;

    int8_t* xp = (int8_t*)d_ws;
    int8_t* wp = xp + (size_t)B_ * HP * WP * CIN;

    {
        const int total = B_ * HP * WP * (CIN / 4);
        pack_input<<<(total + 255) / 256, 256, 0, stream>>>(cast_in, (uint32_t*)xp);
    }
    {
        const int total = 9 * 2 * COUT * 16;
        pack_weight<<<(total + 255) / 256, 256, 0, stream>>>(conv_w, (uint32_t*)wp);
    }
    cimu_conv_fused<<<dim3(7, 28, 32), 256, 0, stream>>>(
        xp, wp, conv_b, merge_in, gather_in,
        ctrans_i, ctrans_s, strans_i, strans_s, atrans_i, atrans_s, out);
}
